// AxialAttentionBlock_49890340110981
// MI455X (gfx1250) — compile-verified, matrix operand order checked by reading, not run on hardware
//
#include <hip/hip_runtime.h>
#include <hip/hip_bf16.h>

// ---------------------------------------------------------------------------
// AxialAttentionBlock on MI455X (gfx1250, wave32, WMMA).
// All GEMM-like work runs through v_wmma_f32_16x16x32_bf16 (bf16 in, f32 acc).
// Generic GEMM stages tiles into LDS with global_load_async_to_lds_b128
// (double-buffered, s_wait_asynccnt) and feeds fragments via ds_load_b128.
// Token-major buffers: token t = n*L + l, feature c; T = N*L = 16384, C = 768.
// Workspace requirement: ~440 MB (see layout in kernel_launch).
// ---------------------------------------------------------------------------

typedef __bf16 bf16;
typedef __attribute__((ext_vector_type(16))) __bf16 v16bf;
typedef __attribute__((ext_vector_type(8)))  __bf16 v8bf;
typedef __attribute__((ext_vector_type(8)))  float   v8f;

#define DEVI __device__ __forceinline__

// A-style 16x32 bf16 fragment load (ISA 7.12.2, 16-bit A layout):
// lanes 0-15 hold row M=lane with K 0..7 (vgpr0-3) and 16..23 (vgpr4-7);
// lanes 16-31 hold row M=lane-16 with K 8..15 and 24..31.
// => per lane: two contiguous 16-byte chunks from one row.
DEVI v16bf load_frag_g(const bf16* __restrict__ base, int ld) {
  int lane = threadIdx.x & 31;
  const bf16* p = base + (size_t)(lane & 15) * (size_t)ld + ((lane >> 4) << 3);
  v8bf lo = *(const v8bf*)p;
  v8bf hi = *(const v8bf*)(p + 16);
  v16bf r;
#pragma unroll
  for (int i = 0; i < 8; ++i) { r[i] = lo[i]; r[i + 8] = hi[i]; }
  return r;
}

// Same fragment pattern but from an LDS tile with row stride LP elements.
template <int LP>
DEVI v16bf load_frag_lds(const bf16* base) {
  int lane = threadIdx.x & 31;
  const bf16* p = base + (lane & 15) * LP + ((lane >> 4) << 3);
  v8bf lo = *(const v8bf*)p;
  v8bf hi = *(const v8bf*)(p + 16);
  v16bf r;
#pragma unroll
  for (int i = 0; i < 8; ++i) { r[i] = lo[i]; r[i + 8] = hi[i]; }
  return r;
}

DEVI v8f wmma_bf16(v16bf a, v16bf b, v8f c) {
  // 8 args: (neg_a, A, neg_b, B, c_mod, C, reuse_a, reuse_b)
  return __builtin_amdgcn_wmma_f32_16x16x32_bf16(false, a, false, b, (short)0, c,
                                                 false, false);
}

// Async global->LDS copy, 16 bytes per lane (ISA 08_async_tensor 4.2/4.4):
// VDST = per-lane LDS byte offset, VADDR = per-lane 64-bit global address.
// Generic LDS pointers carry the LDS byte offset in their low 32 bits.
DEVI void async_g2l_b128(const bf16* lds_ptr, const bf16* gptr) {
  unsigned lds_off = (unsigned)(size_t)lds_ptr;
  asm volatile("global_load_async_to_lds_b128 %0, %1, off"
               :: "v"(lds_off), "v"(gptr)
               : "memory");
}
DEVI void wait_async_le5() { asm volatile("s_wait_asynccnt 5" ::: "memory"); }
DEVI void wait_async_0()   { asm volatile("s_wait_asynccnt 0" ::: "memory"); }

// ---------------------------------------------------------------------------
// Elementwise f32 -> bf16
// ---------------------------------------------------------------------------
__global__ void cvt_bf16_kernel(const float* __restrict__ in, bf16* __restrict__ out,
                                size_t n) {
  size_t i = (size_t)blockIdx.x * 256 + threadIdx.x;
  size_t stride = (size_t)gridDim.x * 256;
  for (; i < n; i += stride) out[i] = (bf16)in[i];
}

// ---------------------------------------------------------------------------
// Transpose f32 (R x Cc) -> bf16 (Cc x R).  block (32,8), grid (Cc/32, R/32)
// ---------------------------------------------------------------------------
__global__ void transpose_cvt_kernel(const float* __restrict__ in,
                                     bf16* __restrict__ out, int R, int Cc) {
  __shared__ float tile[32][33];
  int c0 = blockIdx.x * 32, r0 = blockIdx.y * 32;
  int tx = threadIdx.x, ty = threadIdx.y;
#pragma unroll
  for (int i = 0; i < 32; i += 8)
    tile[ty + i][tx] = in[(size_t)(r0 + ty + i) * Cc + c0 + tx];
  __syncthreads();
#pragma unroll
  for (int i = 0; i < 32; i += 8)
    out[(size_t)(c0 + ty + i) * R + r0 + tx] = (bf16)tile[tx][ty + i];
}

// ---------------------------------------------------------------------------
// Row-wise LayerNorm over C=768, bf16 output.  One 256-thread block per row.
// ---------------------------------------------------------------------------
__global__ void layernorm_bf16_kernel(const float* __restrict__ x,
                                      const float* __restrict__ w,
                                      const float* __restrict__ b,
                                      bf16* __restrict__ out) {
  __shared__ float red[16];
  int row = blockIdx.x, tid = threadIdx.x;
  const float* xr = x + (size_t)row * 768;
  float v0 = xr[tid], v1 = xr[tid + 256], v2 = xr[tid + 512];
  float s = v0 + v1 + v2;
  float ss = v0 * v0 + v1 * v1 + v2 * v2;
#pragma unroll
  for (int o = 16; o > 0; o >>= 1) {
    s += __shfl_xor(s, o, 32);
    ss += __shfl_xor(ss, o, 32);
  }
  int wv = tid >> 5;
  if ((tid & 31) == 0) { red[wv] = s; red[8 + wv] = ss; }
  __syncthreads();
  s = 0.f; ss = 0.f;
#pragma unroll
  for (int i = 0; i < 8; ++i) { s += red[i]; ss += red[8 + i]; }
  float mu = s * (1.0f / 768.0f);
  float var = ss * (1.0f / 768.0f) - mu * mu;
  float rs = rsqrtf(var + 1e-5f);
  bf16* orow = out + (size_t)row * 768;
  orow[tid]       = (bf16)((v0 - mu) * rs * w[tid]       + b[tid]);
  orow[tid + 256] = (bf16)((v1 - mu) * rs * w[tid + 256] + b[tid + 256]);
  orow[tid + 512] = (bf16)((v2 - mu) * rs * w[tid + 512] + b[tid + 512]);
}

// ---------------------------------------------------------------------------
// Generic GEMM: C(MxN,f32) = A(MxK,bf16) * Bt(NxK,bf16)^T (+bias, optional relu)
// block 256 = 8 waves (2x4), wave tile 32x64, workgroup tile 64x256.
// A/B tiles staged into LDS with async global->LDS b128 copies, double
// buffered; fragments read from LDS (padded rows -> conflict-free ds_load_b128),
// all 6 fragments batched before the 8 WMMAs to avoid per-pair dscnt stalls.
// grid (M/64, N/256). M%64==0, N%256==0, K%32==0.
// ---------------------------------------------------------------------------
__global__ void gemm_bt_kernel(const bf16* __restrict__ A, const bf16* __restrict__ Bt,
                               const float* __restrict__ bias, float* __restrict__ C,
                               int M, int N, int K, int relu) {
  constexpr int LP = 40;  // padded LDS row stride (elements): 80B -> no bank conflicts
  __shared__ __align__(16) bf16 sA[2][64 * LP];
  __shared__ __align__(16) bf16 sB[2][256 * LP];
  int tid = threadIdx.x, wave = tid >> 5, lane = tid & 31;
  int wm = wave & 1, wn = wave >> 1;
  int mBlk = blockIdx.x * 64, nBlk = blockIdx.y * 256;
  int m0l = wm * 32, n0l = wn * 64;

  // Stage one 64x32 A tile + 256x32 B tile (20KB) = 1280 x 16B chunks;
  // 256 threads x 5 async instructions, uniform per wave (ASYNCcnt += 5).
  auto stage = [&](int buf, int k0) {
#pragma unroll
    for (int i = 0; i < 4; ++i) {
      int idx = i * 256 + tid;          // 0..1023 -> B tile
      int row = idx >> 2, ch = idx & 3; // 4 chunks of 8 elems per 32-elem row
      async_g2l_b128(&sB[buf][row * LP + ch * 8],
                     Bt + (size_t)(nBlk + row) * K + k0 + ch * 8);
    }
    {
      int row = tid >> 2, ch = tid & 3; // 256 chunks -> A tile (64 rows)
      async_g2l_b128(&sA[buf][row * LP + ch * 8],
                     A + (size_t)(mBlk + row) * K + k0 + ch * 8);
    }
  };

  v8f acc[2][4] = {};
  int nk = K >> 5;
  stage(0, 0);
  int buf = 0;
  for (int ks = 0; ks < nk; ++ks) {
    if (ks + 1 < nk) {
      stage(buf ^ 1, (ks + 1) << 5);
      wait_async_le5();  // in-order completion => previous tile has landed
    } else {
      wait_async_0();
    }
    __syncthreads();
    // batch all fragment loads (12x ds_load_b128), then 8 back-to-back WMMAs
    v16bf a0 = load_frag_lds<LP>(&sA[buf][(m0l + 0) * LP]);
    v16bf a1 = load_frag_lds<LP>(&sA[buf][(m0l + 16) * LP]);
    v16bf bfr[4];
#pragma unroll
    for (int tj = 0; tj < 4; ++tj)
      bfr[tj] = load_frag_lds<LP>(&sB[buf][(n0l + tj * 16) * LP]);
#pragma unroll
    for (int tj = 0; tj < 4; ++tj) {
      acc[0][tj] = wmma_bf16(a0, bfr[tj], acc[0][tj]);
      acc[1][tj] = wmma_bf16(a1, bfr[tj], acc[1][tj]);
    }
    __syncthreads();  // everyone done with buf before it is overwritten
    buf ^= 1;
  }

  // D layout: vgpr r -> row (lane<16 ? r : r+8), col = lane&15
  int col = lane & 15, rb = (lane >> 4) << 3;
#pragma unroll
  for (int ti = 0; ti < 2; ++ti)
#pragma unroll
    for (int tj = 0; tj < 4; ++tj) {
      int nn = nBlk + n0l + tj * 16 + col;
      float bv = bias ? bias[nn] : 0.0f;
#pragma unroll
      for (int r = 0; r < 8; ++r) {
        int mm = mBlk + m0l + ti * 16 + rb + r;
        float v = acc[ti][tj][r] + bv;
        if (relu) v = fmaxf(v, 0.0f);
        C[(size_t)mm * N + nn] = v;
      }
    }
}

// ---------------------------------------------------------------------------
// Row-attention logits: logits[h,i,j] = (1/64) * sum_{n,d} Q[(n,i),hD+d]*K[(n,j),hD+d]
// 256x256 per head, K-dim 64*64=4096.  grid (4,2,12), block 256 (waves 2x4).
// ---------------------------------------------------------------------------
__global__ void row_logits_kernel(const bf16* __restrict__ Qb,
                                  const bf16* __restrict__ Kb,
                                  float* __restrict__ logits) {
  int tid = threadIdx.x, wave = tid >> 5, lane = tid & 31;
  int wm = wave & 1, wn = wave >> 1;
  int h = blockIdx.z;
  int i0 = blockIdx.x * 64 + wm * 32;
  int j0 = blockIdx.y * 128 + wn * 32;
  v8f acc[2][2] = {};
  for (int n = 0; n < 64; ++n) {
    const bf16* qb = Qb + (size_t)(n * 256 + i0) * 768 + h * 64;
    const bf16* kb = Kb + (size_t)(n * 256 + j0) * 768 + h * 64;
#pragma unroll
    for (int kk = 0; kk < 64; kk += 32) {
      v16bf a0 = load_frag_g(qb + kk, 768);
      v16bf a1 = load_frag_g(qb + (size_t)16 * 768 + kk, 768);
      v16bf b0 = load_frag_g(kb + kk, 768);
      v16bf b1 = load_frag_g(kb + (size_t)16 * 768 + kk, 768);
      acc[0][0] = wmma_bf16(a0, b0, acc[0][0]);
      acc[0][1] = wmma_bf16(a0, b1, acc[0][1]);
      acc[1][0] = wmma_bf16(a1, b0, acc[1][0]);
      acc[1][1] = wmma_bf16(a1, b1, acc[1][1]);
    }
  }
  const float sc = 1.0f / 64.0f;  // D^-0.5 / sqrt(R) = (1/8)/8
  int col = lane & 15, rb = (lane >> 4) << 3;
  float* lg = logits + (size_t)h * 256 * 256;
#pragma unroll
  for (int ti = 0; ti < 2; ++ti)
#pragma unroll
    for (int tj = 0; tj < 2; ++tj)
#pragma unroll
      for (int r = 0; r < 8; ++r)
        lg[(size_t)(i0 + ti * 16 + rb + r) * 256 + j0 + tj * 16 + col] =
            acc[ti][tj][r] * sc;
}

// ---------------------------------------------------------------------------
// Softmax over rows of length 256 -> bf16 probs.  One block per row.
// ---------------------------------------------------------------------------
__global__ void softmax_rows_kernel(const float* __restrict__ logits,
                                    bf16* __restrict__ probs) {
  __shared__ float red[8];
  size_t row = blockIdx.x;
  int tid = threadIdx.x;
  float v = logits[row * 256 + tid];
  float m = v;
#pragma unroll
  for (int o = 16; o > 0; o >>= 1) m = fmaxf(m, __shfl_xor(m, o, 32));
  int wv = tid >> 5;
  if ((tid & 31) == 0) red[wv] = m;
  __syncthreads();
  float bm = red[0];
#pragma unroll
  for (int i = 1; i < 8; ++i) bm = fmaxf(bm, red[i]);
  __syncthreads();
  float e = __expf(v - bm);
  float s = e;
#pragma unroll
  for (int o = 16; o > 0; o >>= 1) s += __shfl_xor(s, o, 32);
  if ((tid & 31) == 0) red[wv] = s;
  __syncthreads();
  float bs = 0.f;
#pragma unroll
  for (int i = 0; i < 8; ++i) bs += red[i];
  probs[row * 256 + tid] = (bf16)(e / bs);
}

// ---------------------------------------------------------------------------
// Row-attention context: ctx[(n,i),hD+d] = sum_j probs[h,i,j] * VT[hD+d,(n,j)]
// grid (2, 64(n), 12(h)), block 256 (waves 4x2), wave tile 32(i) x 32(d).
// ---------------------------------------------------------------------------
__global__ void row_ctx_kernel(const bf16* __restrict__ probsb,
                               const bf16* __restrict__ VT,
                               float* __restrict__ ctx) {
  int tid = threadIdx.x, wave = tid >> 5, lane = tid & 31;
  int wm = wave >> 1, wn = wave & 1;
  int h = blockIdx.z, n = blockIdx.y;
  int i0 = blockIdx.x * 128 + wm * 32;
  int d0 = wn * 32;
  v8f acc[2][2] = {};
  const bf16* pb = probsb + (size_t)(h * 256 + i0) * 256;
  const bf16* vb = VT + (size_t)(h * 64 + d0) * 16384 + n * 256;
#pragma unroll 2
  for (int j0 = 0; j0 < 256; j0 += 32) {
    v16bf a0 = load_frag_g(pb + j0, 256);
    v16bf a1 = load_frag_g(pb + (size_t)16 * 256 + j0, 256);
    v16bf b0 = load_frag_g(vb + j0, 16384);
    v16bf b1 = load_frag_g(vb + (size_t)16 * 16384 + j0, 16384);
    acc[0][0] = wmma_bf16(a0, b0, acc[0][0]);
    acc[0][1] = wmma_bf16(a0, b1, acc[0][1]);
    acc[1][0] = wmma_bf16(a1, b0, acc[1][0]);
    acc[1][1] = wmma_bf16(a1, b1, acc[1][1]);
  }
  int col = lane & 15, rb = (lane >> 4) << 3;
#pragma unroll
  for (int ti = 0; ti < 2; ++ti)
#pragma unroll
    for (int tj = 0; tj < 2; ++tj)
#pragma unroll
      for (int r = 0; r < 8; ++r)
        ctx[(size_t)(n * 256 + i0 + ti * 16 + rb + r) * 768 + h * 64 + d0 +
            tj * 16 + col] = acc[ti][tj][r];
}

// ---------------------------------------------------------------------------
// Fused column attention for one (head h, column c, 32-row strip of i):
//   logits (WMMA, global Q/K) -> LDS f32 -> softmax (shfl groups of 8)
//   -> ctx (WMMA, probs from LDS f32->bf16, V^T staged in LDS bf16).
// grid (8, 64, 12), block 256, dynamic LDS = 32*256*4 + 64*256*2 = 64 KB.
// ---------------------------------------------------------------------------
__global__ void col_attn_fused_kernel(const bf16* __restrict__ Qb,
                                      const bf16* __restrict__ Kb,
                                      const float* __restrict__ Vf,
                                      float* __restrict__ ctx) {
  extern __shared__ char smem[];
  float* sl = (float*)smem;                       // [32][256] logits/probs
  bf16*  sv = (bf16*)(smem + 32 * 256 * 4);       // [64][256] V^T slab
  int strip = blockIdx.x, c = blockIdx.y, h = blockIdx.z;
  int tid = threadIdx.x, wave = tid >> 5, lane = tid & 31;
  int i0 = strip * 32;

  // stage V^T: sv[d][j] = V[(c*256+j)*768 + h*64 + d]
  const float* vbase = Vf + (size_t)(c * 256) * 768 + h * 64;
  for (int idx = tid; idx < 64 * 256; idx += 256) {
    int d = idx & 63, j = idx >> 6;
    sv[d * 256 + j] = (bf16)vbase[(size_t)j * 768 + d];
  }

  // logits: wave covers 32 rows x 32 cols (cols j0 = wave*32), K-dim 64
  int j0 = wave * 32;
  v8f acc[2][2] = {};
  const bf16* qb = Qb + (size_t)(c * 256 + i0) * 768 + h * 64;
  const bf16* kb = Kb + (size_t)(c * 256 + j0) * 768 + h * 64;
#pragma unroll
  for (int kk = 0; kk < 64; kk += 32) {
    v16bf a0 = load_frag_g(qb + kk, 768);
    v16bf a1 = load_frag_g(qb + (size_t)16 * 768 + kk, 768);
    v16bf b0 = load_frag_g(kb + kk, 768);
    v16bf b1 = load_frag_g(kb + (size_t)16 * 768 + kk, 768);
    acc[0][0] = wmma_bf16(a0, b0, acc[0][0]);
    acc[0][1] = wmma_bf16(a0, b1, acc[0][1]);
    acc[1][0] = wmma_bf16(a1, b0, acc[1][0]);
    acc[1][1] = wmma_bf16(a1, b1, acc[1][1]);
  }
  {
    int col = lane & 15, rb = (lane >> 4) << 3;
#pragma unroll
    for (int ti = 0; ti < 2; ++ti)
#pragma unroll
      for (int tj = 0; tj < 2; ++tj)
#pragma unroll
        for (int r = 0; r < 8; ++r)
          sl[(ti * 16 + rb + r) * 256 + j0 + tj * 16 + col] =
              acc[ti][tj][r] * 0.125f;  // D^-0.5
  }
  __syncthreads();

  // softmax per row: 8 threads/row (contiguous -> same wave), 32 cols each
  {
    int row = tid >> 3, seg = tid & 7;
    float* rp = sl + row * 256 + seg * 32;
    float m = -3.0e38f;
#pragma unroll
    for (int t = 0; t < 32; ++t) m = fmaxf(m, rp[t]);
#pragma unroll
    for (int msk = 1; msk < 8; msk <<= 1) m = fmaxf(m, __shfl_xor(m, msk, 32));
    float s = 0.f;
#pragma unroll
    for (int t = 0; t < 32; ++t) { float e = __expf(rp[t] - m); rp[t] = e; s += e; }
#pragma unroll
    for (int msk = 1; msk < 8; msk <<= 1) s += __shfl_xor(s, msk, 32);
    float inv = 1.0f / s;
#pragma unroll
    for (int t = 0; t < 32; ++t) rp[t] *= inv;
  }
  __syncthreads();

  // ctx: strip output 32(i) x 64(d) = 8 tiles of 16x16, one per wave. K=256.
  {
    int ti = (wave & 1) * 16, td = (wave >> 1) * 16;
    int lane15 = lane & 15, k8 = (lane >> 4) << 3;
    v8f cacc = {};
#pragma unroll 2
    for (int jj = 0; jj < 256; jj += 32) {
      const float* ap = sl + (ti + lane15) * 256 + jj + k8;
      v16bf a;
#pragma unroll
      for (int t = 0; t < 8; ++t) { a[t] = (bf16)ap[t]; a[t + 8] = (bf16)ap[t + 16]; }
      const bf16* bp = sv + (td + lane15) * 256 + jj + k8;
      v8bf lo = *(const v8bf*)bp;
      v8bf hi = *(const v8bf*)(bp + 16);
      v16bf b;
#pragma unroll
      for (int t = 0; t < 8; ++t) { b[t] = lo[t]; b[t + 8] = hi[t]; }
      cacc = wmma_bf16(a, b, cacc);
    }
    int col = lane & 15, rb = (lane >> 4) << 3;
#pragma unroll
    for (int r = 0; r < 8; ++r)
      ctx[(size_t)(c * 256 + i0 + ti + rb + r) * 768 + h * 64 + td + col] = cacc[r];
  }
}

// ---------------------------------------------------------------------------
// Orchestration
// ---------------------------------------------------------------------------
extern "C" void kernel_launch(void* const* d_in, const int* in_sizes, int n_in,
                              void* d_out, int out_size, void* d_ws, size_t ws_size,
                              hipStream_t stream) {
  (void)in_sizes; (void)n_in; (void)out_size; (void)ws_size;
  const float* x     = (const float*)d_in[0];
  const float* w_proj[8] = {(const float*)d_in[1], (const float*)d_in[2],
                            (const float*)d_in[3], (const float*)d_in[4],
                            (const float*)d_in[5], (const float*)d_in[6],
                            (const float*)d_in[7], (const float*)d_in[8]};
  const float* ln_w[3] = {(const float*)d_in[9],  (const float*)d_in[11],
                          (const float*)d_in[13]};
  const float* ln_b[3] = {(const float*)d_in[10], (const float*)d_in[12],
                          (const float*)d_in[14]};
  const float* ffn_w1 = (const float*)d_in[15];
  const float* ffn_b1 = (const float*)d_in[16];
  const float* ffn_w2 = (const float*)d_in[17];
  const float* ffn_b2 = (const float*)d_in[18];
  float* out = (float*)d_out;

  const size_t T = 16384, C = 768, F = 3072;
  char* ws = (char*)d_ws;
  size_t cur = 0;
  auto alloc = [&](size_t bytes) -> char* {
    char* p = ws + cur;
    cur += (bytes + 255) & ~(size_t)255;
    return p;
  };

  bf16* wT[8];
  for (int i = 0; i < 8; ++i) wT[i] = (bf16*)alloc(C * C * 2);
  bf16* w1T    = (bf16*)alloc(F * C * 2);        // (F x C)
  bf16* w2T    = (bf16*)alloc(C * F * 2);        // (C x F)
  bf16* lnb    = (bf16*)alloc(T * C * 2);
  float* Qf    = (float*)alloc(T * C * 4);
  float* Kf    = (float*)alloc(T * C * 4);
  float* Vf    = (float*)alloc(T * C * 4);
  bf16* Qb     = (bf16*)alloc(T * C * 2);
  bf16* Kb     = (bf16*)alloc(T * C * 2);
  bf16* VT     = (bf16*)alloc(C * T * 2);        // (C x T)
  float* lgR   = (float*)alloc((size_t)12 * 256 * 256 * 4);
  bf16* prbR   = (bf16*)alloc((size_t)12 * 256 * 256 * 2);
  float* ctxf  = (float*)alloc(T * C * 4);
  bf16* ctxb   = (bf16*)alloc(T * C * 2);
  float* aout  = (float*)alloc(T * C * 4);
  float* hf    = (float*)alloc((size_t)2048 * F * 4);
  bf16* hb     = (bf16*)alloc((size_t)2048 * F * 2);

  dim3 tb(32, 8);
  // 1) weight transposes -> bf16 B^T operands
  for (int i = 0; i < 8; ++i)
    transpose_cvt_kernel<<<dim3(C / 32, C / 32), tb, 0, stream>>>(w_proj[i], wT[i],
                                                                  (int)C, (int)C);
  transpose_cvt_kernel<<<dim3(F / 32, C / 32), tb, 0, stream>>>(ffn_w1, w1T, (int)C, (int)F);
  transpose_cvt_kernel<<<dim3(C / 32, F / 32), tb, 0, stream>>>(ffn_w2, w2T, (int)F, (int)C);

  dim3 gM(T / 64, C / 256);  // generic GEMM: workgroup tile 64 x 256
  // 2) LN1 + row Q/K/V projections
  layernorm_bf16_kernel<<<T, 256, 0, stream>>>(x, ln_w[0], ln_b[0], lnb);
  gemm_bt_kernel<<<gM, 256, 0, stream>>>(lnb, wT[0], nullptr, Qf, (int)T, (int)C, (int)C, 0);
  gemm_bt_kernel<<<gM, 256, 0, stream>>>(lnb, wT[1], nullptr, Kf, (int)T, (int)C, (int)C, 0);
  gemm_bt_kernel<<<gM, 256, 0, stream>>>(lnb, wT[2], nullptr, Vf, (int)T, (int)C, (int)C, 0);
  cvt_bf16_kernel<<<4096, 256, 0, stream>>>(Qf, Qb, T * C);
  cvt_bf16_kernel<<<4096, 256, 0, stream>>>(Kf, Kb, T * C);
  transpose_cvt_kernel<<<dim3(C / 32, T / 32), tb, 0, stream>>>(Vf, VT, (int)T, (int)C);

  // 3) row attention (logits summed over the 64 rows, single softmax)
  row_logits_kernel<<<dim3(4, 2, 12), 256, 0, stream>>>(Qb, Kb, lgR);
  softmax_rows_kernel<<<12 * 256, 256, 0, stream>>>(lgR, prbR);
  row_ctx_kernel<<<dim3(2, 64, 12), 256, 0, stream>>>(prbR, VT, ctxf);
  cvt_bf16_kernel<<<4096, 256, 0, stream>>>(ctxf, ctxb, T * C);
  gemm_bt_kernel<<<gM, 256, 0, stream>>>(ctxb, wT[3], nullptr, aout, (int)T, (int)C, (int)C, 0);

  // 4) LN2 + column Q/K/V projections
  layernorm_bf16_kernel<<<T, 256, 0, stream>>>(aout, ln_w[1], ln_b[1], lnb);
  gemm_bt_kernel<<<gM, 256, 0, stream>>>(lnb, wT[4], nullptr, Qf, (int)T, (int)C, (int)C, 0);
  gemm_bt_kernel<<<gM, 256, 0, stream>>>(lnb, wT[5], nullptr, Kf, (int)T, (int)C, (int)C, 0);
  gemm_bt_kernel<<<gM, 256, 0, stream>>>(lnb, wT[6], nullptr, Vf, (int)T, (int)C, (int)C, 0);
  cvt_bf16_kernel<<<4096, 256, 0, stream>>>(Qf, Qb, T * C);
  cvt_bf16_kernel<<<4096, 256, 0, stream>>>(Kf, Kb, T * C);

  // 5) fused column attention (per head x column x 32-row strip), 64 KB LDS
  col_attn_fused_kernel<<<dim3(8, 64, 12), 256, 64 * 1024, stream>>>(Qb, Kb, Vf, ctxf);
  cvt_bf16_kernel<<<4096, 256, 0, stream>>>(ctxf, ctxb, T * C);
  gemm_bt_kernel<<<gM, 256, 0, stream>>>(ctxb, wT[7], nullptr, aout, (int)T, (int)C, (int)C, 0);

  // 6) LN3 + FFN (chunked over tokens to bound scratch)
  layernorm_bf16_kernel<<<T, 256, 0, stream>>>(aout, ln_w[2], ln_b[2], lnb);
  for (int ck = 0; ck < 8; ++ck) {
    const bf16* ain = lnb + (size_t)ck * 2048 * C;
    gemm_bt_kernel<<<dim3(2048 / 64, F / 256), 256, 0, stream>>>(
        ain, w1T, ffn_b1, hf, 2048, (int)F, (int)C, 1);
    cvt_bf16_kernel<<<4096, 256, 0, stream>>>(hf, hb, (size_t)2048 * F);
    gemm_bt_kernel<<<dim3(2048 / 64, C / 256), 256, 0, stream>>>(
        hb, w2T, ffn_b2, out + (size_t)ck * 2048 * C, 2048, (int)C, (int)F, 0);
  }
}
